// FunctionNPairLoss_1864015806540
// MI455X (gfx1250) — matrix operand order repaired and checked by reading, not compiled, not run on hardware
//
#include <hip/hip_runtime.h>
#include <stdint.h>

#define N     8192
#define KD    128
#define NCLS  64
#define NTILE 64      // N / 128 tiles per dimension

typedef __attribute__((ext_vector_type(16))) __bf16   v16bf;
typedef __attribute__((ext_vector_type(8)))  float    v8f;
typedef __attribute__((ext_vector_type(4)))  uint32_t v4u;
typedef __attribute__((ext_vector_type(4)))  uint32_t u32x4;
typedef __attribute__((ext_vector_type(8)))  int      i32x8;
typedef __attribute__((ext_vector_type(4)))  int      i32x4;

union FragU { v4u u[2]; v16bf f; };

__device__ __forceinline__ uint16_t f32_to_bf16_rne(float x) {
  uint32_t u = __float_as_uint(x);
  u += 0x7FFFu + ((u >> 16) & 1u);
  return (uint16_t)(u >> 16);
}

// A-frag, 16-bit A 16x32 layout (ISA 7.12.2): lane L holds row rowBase+(L&15);
// half=L>>4; dwords 0..3 hold K = kk+half*8+{0..7}, dwords 4..7 hold +16.
__device__ __forceinline__ v16bf load_a_frag(const uint16_t* __restrict__ E,
                                             int rowBase, int kk, int lane) {
  const int half = lane >> 4;
  const uint16_t* p = E + (rowBase + (lane & 15)) * KD + kk + half * 8;
  FragU f;
  f.u[0] = *(const v4u*)(p);
  f.u[1] = *(const v4u*)(p + 16);
  return f.f;
}

// B-frag from the LDS-staged 128xKD panel: lane L holds column colLocal+(L&15);
// K = kk+half*8+{0..7} and kk+16+half*8+{0..7} -> two ds_load_b128.
__device__ __forceinline__ v16bf load_b_frag_lds(const uint16_t* Bs,
                                                 int colLocal, int kk, int lane) {
  const int half = lane >> 4;
  const uint16_t* p = Bs + (colLocal + (lane & 15)) * KD + kk + half * 8;
  FragU f;
  f.u[0] = *(const v4u*)(p);
  f.u[1] = *(const v4u*)(p + 16);
  return f.f;
}

// ---- Tensor Data Mover: DMA a 128-row x KD-col bf16 panel (tile start =
// gsrc) into LDS at lds_off. D# per ISA ch.8; 2-D tensor -> groups 2/3 zero.
// This toolchain exposes the 6-arg builtin: (g0, g1, g2, g3, g4, cpol).
__device__ __forceinline__ void tdm_load_panel(const uint16_t* gsrc,
                                               uint32_t lds_off) {
  const uint64_t ga = (uint64_t)(uintptr_t)gsrc;
  u32x4 g0;
  g0[0] = 1u;                                   // count=1 (valid), no gather
  g0[1] = lds_off;                              // lds_addr (bytes)
  g0[2] = (uint32_t)ga;                         // global_addr[31:0]
  g0[3] = (uint32_t)((ga >> 32) & 0x01FFFFFFu)  // global_addr[56:32]
          | (2u << 30);                         // type = 2 ("image")
  i32x8 g1;
  g1[0] = (int)(1u << 16);        // workgroup_mask=0, data_size=1 (2 bytes)
  g1[1] = (int)((uint32_t)KD << 16);   // tensor_dim0[15:0] in bits[63:48]
  g1[2] = (int)((uint32_t)N << 16);    // tensor_dim0[31:16]=0 | tensor_dim1[15:0]
  g1[3] = (int)((uint32_t)KD << 16);   // tensor_dim1[31:16]=0 | tile_dim0=KD
  g1[4] = (int)128;                    // tile_dim1=128 rows, tile_dim2=0
  g1[5] = (int)KD;                     // tensor_dim0_stride[31:0] = KD
  g1[6] = 0;                           // stride0[47:32]=0, stride1[15:0]=0
  g1[7] = 0;
  i32x4 z4 = {0, 0, 0, 0};
  i32x8 z8 = {0, 0, 0, 0, 0, 0, 0, 0};
  __builtin_amdgcn_tensor_load_to_lds(g0, g1, z4, z4, z8, 0);
}

// One wave computes a 16x128 strip of E*E^T: A frags straight from global
// (L2-resident, no redundancy), B frags from the TDM-staged LDS panel.
// B frags grouped 4-at-a-time into distinct registers so the WMMAs issue
// back-to-back after a single wait instead of load->wait->wmma chains.
__device__ __forceinline__ void gemm_tile(const uint16_t* __restrict__ E,
                                          const uint16_t* Bs,
                                          int rowBase, int lane, v8f acc[8]) {
  v8f zero = {};
#pragma unroll
  for (int s = 0; s < 8; ++s) acc[s] = zero;
#pragma unroll
  for (int kk = 0; kk < KD; kk += 32) {
    v16bf a = load_a_frag(E, rowBase, kk, lane);
    v16bf b[4];
#pragma unroll
    for (int s = 0; s < 4; ++s) b[s] = load_b_frag_lds(Bs, s * 16, kk, lane);
#pragma unroll
    for (int s = 0; s < 4; ++s)
      acc[s] = __builtin_amdgcn_wmma_f32_16x16x32_bf16(
          false, a, false, b[s], (short)0, acc[s], false, false);
#pragma unroll
    for (int s = 0; s < 4; ++s) b[s] = load_b_frag_lds(Bs, (s + 4) * 16, kk, lane);
#pragma unroll
    for (int s = 0; s < 4; ++s)
      acc[s + 4] = __builtin_amdgcn_wmma_f32_16x16x32_bf16(
          false, a, false, b[s], (short)0, acc[s + 4], false, false);
  }
}

// ---------------- kernel 1: fp32 -> bf16 + row squared norms ----------------
__global__ void __launch_bounds__(KD)
prep_kernel(const float* __restrict__ emb, uint16_t* __restrict__ Ebf,
            float* __restrict__ sq) {
  const int i = blockIdx.x;
  const int t = threadIdx.x;
  const float x = emb[i * KD + t];
  Ebf[i * KD + t] = f32_to_bf16_rne(x);
  __shared__ float red[KD];
  red[t] = x * x;
  __syncthreads();
  for (int off = KD / 2; off > 0; off >>= 1) {
    if (t < off) red[t] += red[t + off];
    __syncthreads();
  }
  if (t == 0) sq[i] = red[0];
}

// ---------------- kernel 2: Z partials (sum over negatives of exp(D)) -------
__global__ void __launch_bounds__(256)
zpass_kernel(const uint16_t* __restrict__ E, const float* __restrict__ sq,
             const int* __restrict__ labels, float* __restrict__ Zpart) {
  __shared__ uint16_t Bs[128 * KD];
  const int lane = threadIdx.x & 31;
  const int wave = threadIdx.x >> 5;
  const int rowBase = blockIdx.y * 128 + wave * 16;
  const int colBase = blockIdx.x * 128;

  if (threadIdx.x < 32) {  // wave 0 issues the TDM DMA (EXEC ignored by TDM)
    tdm_load_panel(E + (size_t)colBase * KD, (uint32_t)(uintptr_t)Bs);
    __builtin_amdgcn_s_wait_tensorcnt(0);
  }
  __syncthreads();

  v8f acc[8];
  gemm_tile(E, Bs, rowBase, lane, acc);

  const int half = lane >> 4;
  const int n = lane & 15;
  float sqI[8];
  int labI[8];
#pragma unroll
  for (int r = 0; r < 8; ++r) {
    const int i = rowBase + half * 8 + r;
    sqI[r] = sq[i];
    labI[r] = labels[i];
  }
  float rowAcc[8];
#pragma unroll
  for (int r = 0; r < 8; ++r) rowAcc[r] = 0.0f;

#pragma unroll
  for (int s = 0; s < 8; ++s) {
    const int j = colBase + s * 16 + n;
    const float sqJ = sq[j];
    const int labJ = labels[j];
#pragma unroll
    for (int r = 0; r < 8; ++r) {
      const float d = sqI[r] + sqJ - 2.0f * acc[s][r];
      if (labI[r] != labJ) rowAcc[r] += __expf(d);
    }
  }
  // reduce across the 16 lanes of each half (xor masks < 16 stay in-half)
#pragma unroll
  for (int r = 0; r < 8; ++r) {
    float v = rowAcc[r];
    v += __shfl_xor(v, 1, 32);
    v += __shfl_xor(v, 2, 32);
    v += __shfl_xor(v, 4, 32);
    v += __shfl_xor(v, 8, 32);
    if (n == 0) Zpart[(rowBase + half * 8 + r) * NTILE + blockIdx.x] = v;
  }
}

// ---------------- kernel 3: reduce 64 partials per row ----------------------
__global__ void __launch_bounds__(256)
reduce64_kernel(const float* __restrict__ part, float* __restrict__ outv) {
  const int i = blockIdx.x * blockDim.x + threadIdx.x;
  if (i < N) {
    float s = 0.0f;
#pragma unroll
    for (int b = 0; b < NTILE; ++b) s += part[i * NTILE + b];
    outv[i] = s;
  }
}

// ---------------- zero helper (graph-capture safe) --------------------------
__global__ void __launch_bounds__(256)
zero_kernel(float* __restrict__ p, int n) {
  const int i = blockIdx.x * blockDim.x + threadIdx.x;
  if (i < n) p[i] = 0.0f;
}

// ---------------- kernel 4: per-anchor pair-loss partials -------------------
__global__ void __launch_bounds__(256)
lpass_kernel(const uint16_t* __restrict__ E, const float* __restrict__ sq,
             const int* __restrict__ labels, const float* __restrict__ Z,
             float* __restrict__ Spart) {
  if (blockIdx.y > blockIdx.x) return;  // i<j only (uniform: whole block exits)

  __shared__ uint16_t Bs[128 * KD];
  const int lane = threadIdx.x & 31;
  const int wave = threadIdx.x >> 5;
  const int rowBase = blockIdx.y * 128 + wave * 16;
  const int colBase = blockIdx.x * 128;

  if (threadIdx.x < 32) {
    tdm_load_panel(E + (size_t)colBase * KD, (uint32_t)(uintptr_t)Bs);
    __builtin_amdgcn_s_wait_tensorcnt(0);
  }
  __syncthreads();

  v8f acc[8];
  gemm_tile(E, Bs, rowBase, lane, acc);

  const int half = lane >> 4;
  const int n = lane & 15;
  float sqI[8], Zi[8];
  int labI[8];
#pragma unroll
  for (int r = 0; r < 8; ++r) {
    const int i = rowBase + half * 8 + r;
    sqI[r] = sq[i];
    labI[r] = labels[i];
    Zi[r] = Z[i];
  }
  float rowAcc[8];
#pragma unroll
  for (int r = 0; r < 8; ++r) rowAcc[r] = 0.0f;

#pragma unroll
  for (int s = 0; s < 8; ++s) {
    const int j = colBase + s * 16 + n;
    const float sqJ = sq[j];
    const int labJ = labels[j];
#pragma unroll
    for (int r = 0; r < 8; ++r) {
      const int i = rowBase + half * 8 + r;
      if (labI[r] == labJ && i < j) {
        const float d = sqI[r] + sqJ - 2.0f * acc[s][r];
        rowAcc[r] += log1pf(Zi[r] * __expf(-d));
      }
    }
  }
#pragma unroll
  for (int r = 0; r < 8; ++r) {
    float v = rowAcc[r];
    v += __shfl_xor(v, 1, 32);
    v += __shfl_xor(v, 2, 32);
    v += __shfl_xor(v, 4, 32);
    v += __shfl_xor(v, 8, 32);
    if (n == 0) Spart[(rowBase + half * 8 + r) * NTILE + blockIdx.x] = v;
  }
}

// ---------------- kernel 5: segment means + final scalar --------------------
__global__ void __launch_bounds__(256)
finalize_kernel(const int* __restrict__ labels, const float* __restrict__ rowSum,
                float* __restrict__ out) {
  __shared__ float csum[NCLS];
  __shared__ unsigned ccnt[NCLS];
  const int t = threadIdx.x;
  if (t < NCLS) { csum[t] = 0.0f; ccnt[t] = 0u; }
  __syncthreads();
  for (int i = t; i < N; i += blockDim.x) {
    const int c = labels[i];
    atomicAdd(&csum[c], rowSum[i]);   // LDS atomics only
    atomicAdd(&ccnt[c], 1u);
  }
  __syncthreads();
  if (t == 0) {
    float sumMean = 0.0f;
    int nValid = 0;
    for (int c = 0; c < NCLS; ++c) {
      const float m = (float)ccnt[c];
      const float pairs = 0.5f * m * (m - 1.0f);  // = segment_sum(row_cnt)
      if (pairs > 0.0f) { sumMean += csum[c] / pairs; ++nValid; }
    }
    out[0] = sumMean / (float)(nValid > 0 ? nValid : 1);
  }
}

extern "C" void kernel_launch(void* const* d_in, const int* in_sizes, int n_in,
                              void* d_out, int out_size, void* d_ws, size_t ws_size,
                              hipStream_t stream) {
  const float* emb = (const float*)d_in[0];
  const int* labels = (const int*)d_in[1];
  float* out = (float*)d_out;

  // Workspace layout (~6.1 MB total)
  char* ws = (char*)d_ws;
  uint16_t* Ebf = (uint16_t*)ws;                 ws += (size_t)N * KD * sizeof(uint16_t);
  float* sq     = (float*)ws;                    ws += (size_t)N * sizeof(float);
  float* Z      = (float*)ws;                    ws += (size_t)N * sizeof(float);
  float* rowSum = (float*)ws;                    ws += (size_t)N * sizeof(float);
  float* Zpart  = (float*)ws;                    ws += (size_t)N * NTILE * sizeof(float);
  float* Spart  = (float*)ws;

  prep_kernel<<<N, KD, 0, stream>>>(emb, Ebf, sq);

  zero_kernel<<<(N * NTILE + 255) / 256, 256, 0, stream>>>(Spart, N * NTILE);

  dim3 grid(NTILE, NTILE);
  zpass_kernel<<<grid, 256, 0, stream>>>(Ebf, sq, labels, Zpart);
  reduce64_kernel<<<(N + 255) / 256, 256, 0, stream>>>(Zpart, Z);

  lpass_kernel<<<grid, 256, 0, stream>>>(Ebf, sq, labels, Z, Spart);
  reduce64_kernel<<<(N + 255) / 256, 256, 0, stream>>>(Spart, rowSum);

  finalize_kernel<<<1, 256, 0, stream>>>(labels, rowSum, out);
}
